// DifferentiableAstar_7095285973815
// MI455X (gfx1250) — matrix operand order repaired and checked by reading, not compiled, not run on hardware
//
#include <hip/hip_runtime.h>

// ---------------------------------------------------------------------------
// Differentiable A* (neural-astar style) for MI455X / gfx1250.
//  - one wave32 per batch element, whole state resident in LDS (~36KB/WGP slot,
//    LDS is 320KB/WGP so occupancy is not LDS-limited)
//  - expand() 3x3 conv done as R*S*R - S with V_WMMA_F32_16X16X32_F16
//    (exact: all matrix operands are 0/1 indicators, f32 accumulate)
//  - band-matrix operand tiles passed BY VALUE (v16h SSA) so they live in
//    VGPRs for the whole 256-step loop -> no scratch spills
//  - global all_solved coupling resolved with a 3-phase launch plan
// ---------------------------------------------------------------------------

typedef __attribute__((ext_vector_type(16))) _Float16 v16h;
typedef __attribute__((ext_vector_type(8)))  float    v8f;

#define SD        32
#define NC        1024      // SD*SD
#define NB        64        // batch
#define TMAXI     256       // 0.25 * 32 * 32
#define G_RATIO   0.5f
#define TB_FACTOR 0.001f
#define INV_SQRTW 0.17677669529663687f   // 1/sqrt(32)

// Store one 16x16 f32 D-tile (ISA C/D layout) into a row-major LDS array.
__device__ __forceinline__ void store_tile_f16(_Float16* __restrict__ dst,
                                               v8f c, int ti, int tj,
                                               int l15, int hi)
{
#pragma unroll
    for (int r = 0; r < 8; ++r) {
        int row = ti * 16 + r + (hi ? 8 : 0);
        dst[row * SD + (tj * 16 + l15)] = (_Float16)c[r];
    }
}
__device__ __forceinline__ void store_tile_f32(float* __restrict__ dst,
                                               v8f c, int ti, int tj,
                                               int l15, int hi)
{
#pragma unroll
    for (int r = 0; r < 8; ++r) {
        int row = ti * 16 + r + (hi ? 8 : 0);
        dst[row * SD + (tj * 16 + l15)] = c[r];
    }
}

// E = R*S*R   (caller subtracts S per-cell to get the 8-neighborhood expand)
__device__ __forceinline__ void expand_wmma(const _Float16* __restrict__ Sh,
                                            _Float16* __restrict__ Th,
                                            float*    __restrict__ E,
                                            v16h rA0, v16h rA1,
                                            v16h rB0, v16h rB1,
                                            int lane)
{
    const int l15 = lane & 15;
    const int hi  = (lane >= 16) ? 1 : 0;
    const int kb  = hi ? 16 : 0;

    // ---- T = R * S ----
#pragma unroll
    for (int j = 0; j < 2; ++j) {
        // B operand: S[:, j*16 + l15]; K = q + (q&8) + 8*hi (same order as A)
        v16h bS;
        const int n = j * 16 + l15;
#pragma unroll
        for (int q = 0; q < 16; ++q) bS[q] = Sh[(q + (q & 8) + (hi ? 8 : 0)) * SD + n];
        {
            v8f c = {};
            c = __builtin_amdgcn_wmma_f32_16x16x32_f16(false, rA0, false, bS,
                                                       (short)0, c, false, false);
            store_tile_f16(Th, c, 0, j, l15, hi);
        }
        {
            v8f c = {};
            c = __builtin_amdgcn_wmma_f32_16x16x32_f16(false, rA1, false, bS,
                                                       (short)0, c, false, false);
            store_tile_f16(Th, c, 1, j, l15, hi);
        }
    }
    __syncthreads();
    // ---- E = T * R ----
#pragma unroll
    for (int i = 0; i < 2; ++i) {
        // A operand: T row block i (16x32), ISA 16-bit A layout
        v16h aT;
        const int m = i * 16 + l15;
#pragma unroll
        for (int q = 0; q < 16; ++q) {
            int k = q + (q & 8) + (hi ? 8 : 0);
            aT[q] = Th[m * SD + k];
        }
        {
            v8f c = {};
            c = __builtin_amdgcn_wmma_f32_16x16x32_f16(false, aT, false, rB0,
                                                       (short)0, c, false, false);
            store_tile_f32(E, c, i, 0, l15, hi);
        }
        {
            v8f c = {};
            c = __builtin_amdgcn_wmma_f32_16x16x32_f16(false, aT, false, rB1,
                                                       (short)0, c, false, false);
            store_tile_f32(E, c, i, 1, l15, hi);
        }
    }
}

// phase 1: run all TMAXI steps, emit per-(b,t) solved bits to ws.
// phase 3: run ctrl[0] steps, backtrack ctrl[1] hops, write hist+path.
__global__ __launch_bounds__(32)
void astar_sim_kernel(const float* __restrict__ costg,
                      const float* __restrict__ startg,
                      const float* __restrict__ goalg,
                      const float* __restrict__ obstg,
                      float* __restrict__ out_hist,
                      float* __restrict__ out_path,
                      unsigned int* __restrict__ solved_bits,
                      const int* __restrict__ ctrl,
                      int phase)
{
    __shared__ float sh_cost[NC], sh_htot[NC], sh_g[NC], sh_open[NC],
                     sh_hist[NC], sh_obst[NC], sh_par[NC], sh_E[NC];
    __shared__ _Float16 sh_S[NC], sh_T[NC];
    __shared__ float red_v[32];
    __shared__ int   red_i[32];
    __shared__ int   sh_gi;

    const int b    = blockIdx.x;
    const int lane = threadIdx.x;
    const int base = b * NC;

    __builtin_prefetch(costg + base, 0, 1);   // global_prefetch_b8
    __builtin_prefetch(obstg + base, 0, 1);

    if (lane == 0) sh_gi = 0;
    __syncthreads();

    // load maps, init state (32 cells per lane, contiguous => argmax ordering)
    for (int c = 0; c < 32; ++c) {
        int i = lane * 32 + c;
        float gv = goalg[base + i];
        sh_cost[i] = costg[base + i];
        sh_open[i] = startg[base + i];
        sh_obst[i] = obstg[base + i];
        sh_g[i]    = 0.0f;
        sh_hist[i] = 0.0f;
        if (gv > 0.5f) sh_gi = i;             // goal map is one-hot
    }
    __syncthreads();
    const int gi = sh_gi;
    const float gy = (float)(gi / SD), gx = (float)(gi % SD);

    for (int c = 0; c < 32; ++c) {
        int i = lane * 32 + c;
        float y = (float)(i / SD), x = (float)(i % SD);
        float dy = fabsf(y - gy), dx = fabsf(x - gx);
        float h = (dy + dx) - fminf(dy, dx)
                + TB_FACTOR * sqrtf(dy * dy + dx * dx + 1e-10f);
        sh_htot[i] = h + sh_cost[i];          // reference: h = heuristic + cost
        sh_par[i]  = (float)gi;               // parents0 = goal index
    }
    __syncthreads();

    // precompute band-matrix R operand tiles as individual SSA vectors
    v16h rA0, rA1, rB0, rB1;
    {
        const int l15 = lane & 15;
        const int hi  = (lane >= 16) ? 1 : 0;
#pragma unroll
        for (int q = 0; q < 16; ++q) {
            int k  = q + (q & 8) + (hi ? 8 : 0);   // A-layout K index
            int d0 = l15 - k;                       // row block 0: m = l15
            int d1 = (16 + l15) - k;                // row block 1: m = 16+l15
            rA0[q] = (_Float16)((d0 >= -1 && d0 <= 1) ? 1.0f : 0.0f);
            rA1[q] = (_Float16)((d1 >= -1 && d1 <= 1) ? 1.0f : 0.0f);
        }
#pragma unroll
        for (int q = 0; q < 16; ++q) {
            int k  = q + (q & 8) + (hi ? 8 : 0);   // B-layout K index
            int d0 = k - l15;                       // col block 0: n = l15
            int d1 = k - (16 + l15);                // col block 1: n = 16+l15
            rB0[q] = (_Float16)((d0 >= -1 && d0 <= 1) ? 1.0f : 0.0f);
            rB1[q] = (_Float16)((d1 >= -1 && d1 <= 1) ? 1.0f : 0.0f);
        }
    }

    const int U = (phase == 1) ? TMAXI : ctrl[0];
    unsigned int curw = 0;

    for (int t = 0; t < U; ++t) {
        // ---- argmax of f_exp (first-max tie-break, like jnp.argmax) ----
        float bv = -1.0f; int bi = 0;
        for (int c = 0; c < 32; ++c) {
            int i = lane * 32 + c;
            float f  = G_RATIO * sh_g[i] + (1.0f - G_RATIO) * sh_htot[i];
            float fe = expf(-f * INV_SQRTW) * sh_open[i];
            if (fe > bv) { bv = fe; bi = i; }
        }
        red_v[lane] = bv; red_i[lane] = bi;
        __syncthreads();
        float mv = -1.0f; int m = 0;
        for (int l = 0; l < 32; ++l)
            if (red_v[l] > mv) { mv = red_v[l]; m = red_i[l]; }

        const float unsolved = (m == gi) ? 0.0f : 1.0f;
        if (phase == 1) {
            curw |= ((m == gi) ? 1u : 0u) << (t & 31);
            if ((t & 31) == 31) {
                if (lane == 0) solved_bits[b * 8 + (t >> 5)] = curw;
                curw = 0;
            }
        }
        const float gval = sh_g[m] + sh_cost[m];   // old g at selected cell

        // ---- sel one-hot (f16, exact) ----
        for (int c = 0; c < 32; ++c) {
            int i = lane * 32 + c;
            sh_S[i] = (_Float16)((i == m) ? 1.0f : 0.0f);
        }
        __syncthreads();

        expand_wmma(sh_S, sh_T, sh_E, rA0, rA1, rB0, rB1, lane); // sh_E = R*S*R
        __syncthreads();

        // ---- dense per-cell update (matches reference op order) ----
        for (int c = 0; c < 32; ++c) {
            int i = lane * 32 + c;
            float sel = (i == m) ? 1.0f : 0.0f;
            float hn  = fminf(sh_hist[i] + sel, 1.0f);
            sh_hist[i] = hn;
            float oa  = fminf(fmaxf(sh_open[i] - unsolved * sel, 0.0f), 1.0f);
            float e   = sh_E[i] - sel;            // expand(sel), exact 0/1
            float nb  = e * sh_obst[i];
            float g2  = gval * e;                 // == conv((g+cost)*sel)
            float gold = sh_g[i];
            float gcmp = (gold > g2) ? 1.0f : 0.0f;
            float idx  = ((1.0f - oa) * (1.0f - hn) + oa * gcmp) * nb;
            sh_g[i]    = g2 * idx + gold * (1.0f - idx);
            sh_open[i] = fminf(oa + idx, 1.0f);
            sh_par[i]  = (float)m * idx + sh_par[i] * (1.0f - idx);
        }
        __syncthreads();
    }

    if (phase == 3) {
        // backtrack: path starts as goal one-hot (reuse sh_E)
        for (int c = 0; c < 32; ++c) {
            int i = lane * 32 + c;
            sh_E[i] = (i == gi) ? 1.0f : 0.0f;
        }
        __syncthreads();
        if (lane == 0) {
            int t_stop = ctrl[1];
            int loc = (int)sh_par[gi];
            for (int s = 0; s < TMAXI; ++s) {
                if (s < t_stop) { sh_E[loc] = 1.0f; loc = (int)sh_par[loc]; }
            }
        }
        __syncthreads();
        for (int c = 0; c < 32; ++c) {
            int i = lane * 32 + c;
            out_hist[base + i] = sh_hist[i];
            out_path[base + i] = sh_E[i];
        }
    }
}

// phase 2: t* = min{ t : all batches solved at t } ; t_stop = min(t*, TMAX-1)
__global__ __launch_bounds__(256)
void astar_reduce_kernel(const unsigned int* __restrict__ solved_bits,
                         int* __restrict__ ctrl)
{
    __shared__ int firstT;
    const int t = threadIdx.x;
    if (t == 0) firstT = TMAXI;
    __syncthreads();
    int all = 1;
    const int w = t >> 5, sb = t & 31;
    for (int b = 0; b < NB; ++b)
        all &= (int)((solved_bits[b * 8 + w] >> sb) & 1u);
    if (all) atomicMin(&firstT, t);
    __syncthreads();
    if (t == 0) {
        int U = firstT;                       // steps to apply (256 if never)
        ctrl[0] = U;
        ctrl[1] = (U < TMAXI - 1) ? U : (TMAXI - 1);
    }
}

extern "C" void kernel_launch(void* const* d_in, const int* in_sizes, int n_in,
                              void* d_out, int out_size, void* d_ws, size_t ws_size,
                              hipStream_t stream)
{
    const float* cost  = (const float*)d_in[0];   // (64,1,32,32)
    const float* start = (const float*)d_in[1];
    const float* goal  = (const float*)d_in[2];
    const float* obst  = (const float*)d_in[3];
    // d_in[4] = neighbor_filter (3x3 ones, zero center) -> hardcoded as R*S*R - S

    float* out_hist = (float*)d_out;
    float* out_path = out_hist + NB * NC;

    unsigned int* solved = (unsigned int*)d_ws;           // NB * 8 words = 2 KB
    int*          ctrl   = (int*)(solved + NB * 8);       // 2 ints

    astar_sim_kernel<<<NB, 32, 0, stream>>>(cost, start, goal, obst,
                                            nullptr, nullptr, solved, nullptr, 1);
    astar_reduce_kernel<<<1, 256, 0, stream>>>(solved, ctrl);
    astar_sim_kernel<<<NB, 32, 0, stream>>>(cost, start, goal, obst,
                                            out_hist, out_path, solved, ctrl, 3);
}
